// CAT_Transformer_58145267253282
// MI455X (gfx1250) — matrix operand order repaired and checked by reading, not compiled, not run on hardware
//
#include <hip/hip_runtime.h>
#include <hip/hip_bf16.h>

// ---------------------------------------------------------------------------
// Types for CDNA5 WMMA (wave32): 16x16x32 f16 -> f32
// ---------------------------------------------------------------------------
typedef __attribute__((ext_vector_type(16))) _Float16 v16h_t;
typedef __attribute__((ext_vector_type(8)))  _Float16 v8h_t;
typedef __attribute__((ext_vector_type(4)))  _Float16 v4h_t;
typedef __attribute__((ext_vector_type(8)))  float    v8f_t;

union Frag16 { v16h_t v; v8h_t h[2]; };

#define GTM 128
#define GTN 64
#define GTK 64   // two WMMA K-steps (2 x 32) per barrier round

// ---------------------------------------------------------------------------
// Generic GEMM:  C[M,N] = A[M,K] @ B[K,N]  (+ compile-time residual R)
// Call-site guarantees: K % 64 == 0, N % 64 == 0, 16B-aligned bases/strides.
// M may be ragged (2178): A rows are CLAMPED to M-1 during staging (tile row r
// only feeds C row r, discarded at store when r >= M) -> branch-free staging:
//   global_load_b128 (float4) -> cvt -> ds_store_b64 / packed b16 stores.
// Block tile 128x64, 8 waves, each wave 32x32 (2x2 WMMA tiles), K-step 64.
// ---------------------------------------------------------------------------
template <bool HAS_R>
__global__ void __launch_bounds__(256)
gemm_wmma_kernel(const float* __restrict__ A, long lda,
                 const float* __restrict__ B, long ldb,
                 float* __restrict__ C, long ldc,
                 const float* __restrict__ R,
                 int M, int N, int Kd)
{
    __shared__ _Float16 As[GTM][GTK];    // row-major A tile (f16)        16KB
    __shared__ _Float16 BsT[GTN][GTK];   // transposed B tile: BsT[n][k]   8KB

    const int tid  = threadIdx.x;
    const int wave = tid >> 5;
    const int lane = tid & 31;
    const int m0 = blockIdx.y * GTM;
    const int n0 = blockIdx.x * GTN;
    const int wm = (wave & 3) * 32;      // wave row offset in tile
    const int wn = (wave >> 2) * 32;     // wave col offset in tile

    // Staging coordinates (branch-free; A rows clamped)
    const int sr  = tid >> 4;            // 0..15 (row group, step 16)
    const int sc4 = tid & 15;            // float4 slot within 64-wide dim

    v8f_t acc[2][2] = {};

    for (int k0 = 0; k0 < Kd; k0 += GTK) {
        if (k0 + GTK < Kd) {
            __builtin_prefetch(&B[(long)(k0 + GTK) * ldb + n0], 0, 1);
        }
        // ---- A tile: 128x64 fp32 -> f16 (8 x float4 per thread) ----
        #pragma unroll
        for (int j = 0; j < 8; ++j) {
            int r  = sr + j * 16;
            int gr = m0 + r; if (gr > M - 1) gr = M - 1;
            const float4 av = *(const float4*)&A[(long)gr * lda + k0 + sc4 * 4];
            v4h_t hv = { (_Float16)av.x, (_Float16)av.y,
                         (_Float16)av.z, (_Float16)av.w };
            *(v4h_t*)&As[r][sc4 * 4] = hv;
        }
        // ---- B tile: 64x64 fp32 -> f16, transposed (4 x float4 per thread) ----
        #pragma unroll
        for (int j = 0; j < 4; ++j) {
            int kr = sr + j * 16;
            const float4 bv = *(const float4*)&B[(long)(k0 + kr) * ldb + n0 + sc4 * 4];
            BsT[sc4 * 4 + 0][kr] = (_Float16)bv.x;
            BsT[sc4 * 4 + 1][kr] = (_Float16)bv.y;
            BsT[sc4 * 4 + 2][kr] = (_Float16)bv.z;
            BsT[sc4 * 4 + 3][kr] = (_Float16)bv.w;
        }
        __syncthreads();

        // ---- fragment loads (contiguous b128 LDS reads) + 8 WMMAs ----
        // A 16x32 layout: lanes 0-15 -> K 0..7/16..23 ; lanes 16-31 -> K 8..15/24..31
        // B 32x16 layout: lanes 0-15 -> K 0..7/16..23 ; lanes 16-31 -> K 8..15/24..31
        const int mrow = lane & 15;
        const int ksel = (lane >> 4) * 8;
        #pragma unroll
        for (int kk = 0; kk < GTK; kk += 32) {
            Frag16 afr[2], bfr[2];
            #pragma unroll
            for (int i = 0; i < 2; ++i) {
                const _Float16* ap = &As[wm + i * 16 + mrow][kk];
                afr[i].h[0] = *(const v8h_t*)(ap + ksel);
                afr[i].h[1] = *(const v8h_t*)(ap + 16 + ksel);
                const _Float16* bp = &BsT[wn + i * 16 + mrow][kk];
                bfr[i].h[0] = *(const v8h_t*)(bp + ksel);
                bfr[i].h[1] = *(const v8h_t*)(bp + 16 + ksel);
            }
            #pragma unroll
            for (int i = 0; i < 2; ++i)
                #pragma unroll
                for (int j = 0; j < 2; ++j)
                    acc[i][j] = __builtin_amdgcn_wmma_f32_16x16x32_f16(
                        false, afr[i].v, false, bfr[j].v,
                        (short)0, acc[i][j], false, false);
        }
        __syncthreads();
    }

    // ---- store C (16x16 f32 D layout: VGPR r -> M=r (+8 hi lanes), N=lane&15) ----
    const int hi8 = (lane >> 4) * 8;
    const int cn  = lane & 15;
    const bool fullM = (m0 + GTM) <= M;
    #pragma unroll
    for (int i = 0; i < 2; ++i)
        #pragma unroll
        for (int j = 0; j < 2; ++j) {
            const int rbase = m0 + wm + i * 16 + hi8;
            const int cc = n0 + wn + j * 16 + cn;
            const long off0 = (long)rbase * ldc + cc;
            if (fullM) {
                float rv[8];
                if (HAS_R) {
                    #pragma unroll
                    for (int r = 0; r < 8; ++r) rv[r] = R[off0 + (long)r * ldc];
                }
                #pragma unroll
                for (int r = 0; r < 8; ++r) {
                    float v = acc[i][j][r];
                    if (HAS_R) v += rv[r];
                    C[off0 + (long)r * ldc] = v;
                }
            } else {
                #pragma unroll
                for (int r = 0; r < 8; ++r) {
                    if (rbase + r < M) {
                        float v = acc[i][j][r];
                        if (HAS_R) v += R[off0 + (long)r * ldc];
                        C[off0 + (long)r * ldc] = v;
                    }
                }
            }
        }
}

// ---------------------------------------------------------------------------
// RMSNorm: one 256-thread block per 1024-wide row
// ---------------------------------------------------------------------------
__global__ void __launch_bounds__(256)
rmsnorm_kernel(const float* __restrict__ x, const float* __restrict__ w,
               float* __restrict__ y)
{
    const int row = blockIdx.x;
    const float* xr = x + (long)row * 1024;
    float s = 0.f;
    for (int i = threadIdx.x; i < 1024; i += 256) { float v = xr[i]; s += v * v; }
    for (int off = 16; off > 0; off >>= 1) s += __shfl_xor(s, off, 32);
    __shared__ float red[8];
    int wv = threadIdx.x >> 5, lane = threadIdx.x & 31;
    if (lane == 0) red[wv] = s;
    __syncthreads();
    if (wv == 0) {
        float t = (lane < 8) ? red[lane] : 0.f;
        for (int off = 4; off > 0; off >>= 1) t += __shfl_xor(t, off, 32);
        if (lane == 0) red[0] = rsqrtf(t / 1024.f + 1e-5f);
    }
    __syncthreads();
    float rs = red[0];
    float* yr = y + (long)row * 1024;
    for (int i = threadIdx.x; i < 1024; i += 256) yr[i] = xr[i] * rs * w[i];
}

// ---------------------------------------------------------------------------
// RoPE (HD=64, half=32). One thread per (row, head, j) pair; in-place safe.
// pos p = (row % rowsPerBatch) % 17
// ---------------------------------------------------------------------------
__global__ void __launch_bounds__(256)
rope_kernel(float* __restrict__ x, int T, int rowsPerBatch)
{
    long idx = (long)blockIdx.x * 256 + threadIdx.x;
    long total = (long)T * 512;
    if (idx >= total) return;
    int j = idx & 31;
    int h = (int)((idx >> 5) & 15);
    int row = (int)(idx >> 9);
    int p = (row % rowsPerBatch) % 17;
    float theta = __powf(10000.f, -(float)j / 32.f);
    float sn, c;
    __sincosf((float)p * theta, &sn, &c);
    float* base = x + (long)row * 1024 + h * 64;
    float x1 = base[j], x2 = base[j + 32];
    base[j]      = x1 * c - x2 * sn;
    base[j + 32] = x2 * c + x1 * sn;
}

// ---------------------------------------------------------------------------
// Fine attention: 128 seqs x 16 heads x L=17 queries, full (no mask).
// One wave per (seq, head, q). Wave32 shuffle softmax.
// ---------------------------------------------------------------------------
__global__ void __launch_bounds__(256)
attn_fine_kernel(const float* __restrict__ Q, const float* __restrict__ K,
                 const float* __restrict__ V, float* __restrict__ O)
{
    int gw = blockIdx.x * 8 + (threadIdx.x >> 5);
    int lane = threadIdx.x & 31;
    if (gw >= 128 * 16 * 17) return;
    int q = gw % 17; int t = gw / 17; int h = t & 15; int s = t >> 4;
    long qrow = (long)(s * 17 + q) * 1024 + h * 64;
    float sc = -1e30f;
    if (lane < 17) {
        long krow = (long)(s * 17 + lane) * 1024 + h * 64;
        float d = 0.f;
        for (int i = 0; i < 64; ++i) d += Q[qrow + i] * K[krow + i];
        sc = d * 0.125f;
    }
    float mx = sc;
    for (int off = 16; off > 0; off >>= 1) mx = fmaxf(mx, __shfl_xor(mx, off, 32));
    float p = (lane < 17) ? __expf(sc - mx) : 0.f;
    float sum = p;
    for (int off = 16; off > 0; off >>= 1) sum += __shfl_xor(sum, off, 32);
    float o0 = 0.f, o1 = 0.f;
    for (int ki = 0; ki < 17; ++ki) {
        float wgt = __shfl(p, ki, 32);
        long vrow = (long)(s * 17 + ki) * 1024 + h * 64;
        o0 += wgt * V[vrow + lane];
        o1 += wgt * V[vrow + 32 + lane];
    }
    float inv = 1.f / sum;
    O[qrow + lane]      = o0 * inv;
    O[qrow + 32 + lane] = o1 * inv;
}

// ---------------------------------------------------------------------------
// Coarse attention with CAT mask. S=1089, block=17.
// Allowed keys for query q: anchors j*17 (j < q/17) + same-block keys [blk..q].
// <= 64 + 17 = 81 candidates -> 3 score registers per lane.
// ---------------------------------------------------------------------------
__global__ void __launch_bounds__(256)
attn_coarse_kernel(const float* __restrict__ Q, const float* __restrict__ K,
                   const float* __restrict__ V, float* __restrict__ O)
{
    const int S = 1089;
    int gw = blockIdx.x * 8 + (threadIdx.x >> 5);
    int lane = threadIdx.x & 31;
    if (gw >= 2 * 16 * S) return;
    int q = gw % S; int t = gw / S; int h = t & 15; int b = t >> 4;
    int qb = q / 17;
    int nc = qb + (q - qb * 17) + 1;
    long base = (long)b * S * 1024;
    long qrow = base + (long)q * 1024 + h * 64;

    float s0 = -1e30f, s1 = -1e30f, s2 = -1e30f;
    for (int c = lane; c < nc; c += 32) {
        int kv = (c < qb) ? c * 17 : qb * 17 + (c - qb);
        long krow = base + (long)kv * 1024 + h * 64;
        float d = 0.f;
        for (int i = 0; i < 64; ++i) d += Q[qrow + i] * K[krow + i];
        float v = d * 0.125f;
        int r = c >> 5;
        if (r == 0) s0 = v; else if (r == 1) s1 = v; else s2 = v;
    }
    float mx = fmaxf(fmaxf(s0, s1), s2);
    for (int off = 16; off > 0; off >>= 1) mx = fmaxf(mx, __shfl_xor(mx, off, 32));
    float p0 = __expf(s0 - mx), p1 = __expf(s1 - mx), p2 = __expf(s2 - mx);
    float sum = p0 + p1 + p2;
    for (int off = 16; off > 0; off >>= 1) sum += __shfl_xor(sum, off, 32);

    float o0 = 0.f, o1 = 0.f;
    for (int c = 0; c < nc; ++c) {
        float pr = (c < 32) ? p0 : ((c < 64) ? p1 : p2);
        float wgt = __shfl(pr, c & 31, 32);
        int kv = (c < qb) ? c * 17 : qb * 17 + (c - qb);
        long vrow = base + (long)kv * 1024 + h * 64;
        o0 += wgt * V[vrow + lane];
        o1 += wgt * V[vrow + 32 + lane];
    }
    float inv = 1.f / sum;
    O[qrow + lane]      = o0 * inv;
    O[qrow + 32 + lane] = o1 * inv;
}

// ---------------------------------------------------------------------------
// g = silu(g) * u
// ---------------------------------------------------------------------------
__global__ void __launch_bounds__(256)
silu_mul_kernel(float* __restrict__ g, const float* __restrict__ u, long n)
{
    long i = (long)blockIdx.x * 256 + threadIdx.x;
    if (i < n) {
        float x = g[i];
        g[i] = (x / (1.f + __expf(-x))) * u[i];
    }
}

// ---------------------------------------------------------------------------
// Fine input assembly: X[(b*64+k)*17 + 0] = pos[k]; +1+t = f.wte[ids[b][k*16+t]]
// ---------------------------------------------------------------------------
__global__ void __launch_bounds__(256)
embed_fine_kernel(const int* __restrict__ ids, const float* __restrict__ fwte,
                  const float* __restrict__ pos, float* __restrict__ X)
{
    long i = (long)blockIdx.x * 256 + threadIdx.x;
    if (i >= (long)2176 * 1024) return;
    int d = i & 1023;
    int row = (int)(i >> 10);
    int t = row % 17; int s = row / 17; int k = s & 63; int b = s >> 6;
    float v;
    if (t == 0) v = pos[k * 1024 + d];
    else        v = fwte[(long)ids[b * 1024 + k * 16 + (t - 1)] * 1024 + d];
    X[i] = v;
}

// ---------------------------------------------------------------------------
// Coarse input assembly (S=1089 rows per batch):
//   token k*17+0 : k==0 ? dummy : fx[b][k-1]
//   token k*17+t : wte[ids[b][k*16+t-1]]   (t=1..16)
//   token 1088   : fx[b][63]
// ---------------------------------------------------------------------------
__global__ void __launch_bounds__(256)
build_coarse_kernel(const int* __restrict__ ids, const float* __restrict__ wte,
                    const float* __restrict__ fx, const float* __restrict__ dummy,
                    float* __restrict__ X)
{
    long i = (long)blockIdx.x * 256 + threadIdx.x;
    if (i >= (long)2178 * 1024) return;
    int d = i & 1023;
    long row = i >> 10;
    int b = (row >= 1089) ? 1 : 0;
    int tok = (int)(row - (long)b * 1089);
    float v;
    if (tok == 1088) v = fx[((long)b * 64 + 63) * 1024 + d];
    else {
        int k = tok / 17, t = tok % 17;
        if (t == 0) v = (k == 0) ? dummy[d] : fx[((long)b * 64 + (k - 1)) * 1024 + d];
        else        v = wte[(long)ids[b * 1024 + k * 16 + (t - 1)] * 1024 + d];
    }
    X[i] = v;
}

// ---------------------------------------------------------------------------
// Gather rows for logits: x_first (tok 1..15), x_middle (k=1..63, t=0..15), x_last
// ---------------------------------------------------------------------------
__global__ void __launch_bounds__(256)
gather_final_kernel(const float* __restrict__ XN, float* __restrict__ XG)
{
    long i = (long)blockIdx.x * 256 + threadIdx.x;
    if (i >= (long)2048 * 1024) return;
    int d = i & 1023;
    long r = i >> 10;
    int b = (r >= 1024) ? 1 : 0;
    int j = (int)(r - (long)b * 1024);
    int tok;
    if (j < 15) tok = 1 + j;
    else if (j < 1023) { int m = j - 15; tok = (m / 16 + 1) * 17 + (m % 16); }
    else tok = 1088;
    XG[i] = XN[((long)b * 1089 + tok) * 1024 + d];
}

// ---------------------------------------------------------------------------
// Host orchestration
// ---------------------------------------------------------------------------
static inline void gemm(const float* A, long lda, const float* B, long ldb,
                        float* C, long ldc, const float* R,
                        int M, int N, int Kd, hipStream_t s)
{
    dim3 grid((N + GTN - 1) / GTN, (M + GTM - 1) / GTM);
    if (R)
        gemm_wmma_kernel<true><<<grid, 256, 0, s>>>(A, lda, B, ldb, C, ldc, R, M, N, Kd);
    else
        gemm_wmma_kernel<false><<<grid, 256, 0, s>>>(A, lda, B, ldb, C, ldc, nullptr, M, N, Kd);
}

struct LayerP { const float *n1, *n2, *wd, *wg, *wk, *wo, *wq, *wu, *wv; };

extern "C" void kernel_launch(void* const* d_in, const int* in_sizes, int n_in,
                              void* d_out, int out_size, void* d_ws, size_t ws_size,
                              hipStream_t stream)
{
    (void)in_sizes; (void)n_in; (void)out_size;
    // jax pytree (sorted-key) flatten order:
    // 0 input_ids, 1 dummy,
    // f.layers[0]: 2 n1,3 n2,4 wd,5 wg,6 wk,7 wo,8 wq,9 wu,10 wv
    // f.layers[1]: 11..19, 20 f.norm, 21 f.pos, 22 f.proj_fx, 23 f.wte
    // layers[0]: 24..32, layers[1]: 33..41, 42 norm, 43 out, 44 wte
    auto in = [&](int i) { return (const float*)d_in[i]; };
    const int*   ids   = (const int*)d_in[0];
    const float* dummy = in(1);
    auto mk = [&](int b0) {
        return LayerP{ in(b0), in(b0 + 1), in(b0 + 2), in(b0 + 3), in(b0 + 4),
                       in(b0 + 5), in(b0 + 6), in(b0 + 7), in(b0 + 8) };
    };
    LayerP fl[2] = { mk(2), mk(11) };
    const float* fnorm = in(20);
    const float* fpos  = in(21);
    const float* fproj = in(22);
    const float* fwte  = in(23);
    LayerP cl[2] = { mk(24), mk(33) };
    const float* cnorm = in(42);
    const float* outw  = in(43);
    const float* wte   = in(44);
    float* logits = (float*)d_out;

    // workspace layout (fp32)
    const long TC = 2178;                 // max tokens (coarse)
    float* W  = (float*)d_ws;
    float* X  = W;
    float* XN = X  + TC * 1024;
    float* Qb = XN + TC * 1024;
    float* Kb = Qb + TC * 1024;
    float* Vb = Kb + TC * 1024;
    float* AO = Vb + TC * 1024;
    float* G  = AO + TC * 1024;
    float* U  = G  + TC * 4096;
    float* FX = U  + TC * 4096;
    float* XG = FX + 2 * 64 * 1024;
    size_t need = (size_t)((XG + (long)2048 * 1024) - W) * sizeof(float);
    if (ws_size < need) return;

    auto run_layer = [&](const LayerP& lp, long T, int ropeMod, bool coarse) {
        rmsnorm_kernel<<<(unsigned)T, 256, 0, stream>>>(X, lp.n1, XN);
        gemm(XN, 1024, lp.wq, 1024, Qb, 1024, nullptr, (int)T, 1024, 1024, stream);
        gemm(XN, 1024, lp.wk, 1024, Kb, 1024, nullptr, (int)T, 1024, 1024, stream);
        gemm(XN, 1024, lp.wv, 1024, Vb, 1024, nullptr, (int)T, 1024, 1024, stream);
        long rt = T * 512;
        rope_kernel<<<(unsigned)((rt + 255) / 256), 256, 0, stream>>>(Qb, (int)T, ropeMod);
        rope_kernel<<<(unsigned)((rt + 255) / 256), 256, 0, stream>>>(Kb, (int)T, ropeMod);
        if (!coarse)
            attn_fine_kernel<<<(128 * 16 * 17) / 8, 256, 0, stream>>>(Qb, Kb, Vb, AO);
        else
            attn_coarse_kernel<<<(2 * 16 * 1089) / 8, 256, 0, stream>>>(Qb, Kb, Vb, AO);
        gemm(AO, 1024, lp.wo, 1024, X, 1024, /*R=*/X, (int)T, 1024, 1024, stream);
        rmsnorm_kernel<<<(unsigned)T, 256, 0, stream>>>(X, lp.n2, XN);
        gemm(XN, 1024, lp.wg, 4096, G, 4096, nullptr, (int)T, 4096, 1024, stream);
        gemm(XN, 1024, lp.wu, 4096, U, 4096, nullptr, (int)T, 4096, 1024, stream);
        long nsu = T * 4096;
        silu_mul_kernel<<<(unsigned)((nsu + 255) / 256), 256, 0, stream>>>(G, U, nsu);
        gemm(G, 4096, lp.wd, 1024, X, 1024, /*R=*/X, (int)T, 1024, 4096, stream);
    };

    // ---- fine stage: 128 seqs x 17 tokens ----
    {
        long n = (long)2176 * 1024;
        embed_fine_kernel<<<(unsigned)((n + 255) / 256), 256, 0, stream>>>(ids, fwte, fpos, X);
    }
    for (int l = 0; l < 2; ++l) run_layer(fl[l], 2176, 17, false);
    rmsnorm_kernel<<<2176, 256, 0, stream>>>(X, fnorm, XN);
    // fx = normed[:,1:,:].reshape(128, 16384) @ proj_fx   (row stride 17*1024, skip token 0)
    gemm(XN + 1024, 17 * 1024, fproj, 1024, FX, 1024, nullptr, 128, 1024, 16384, stream);

    // ---- coarse stage: 2 seqs x 1089 tokens ----
    {
        long n = (long)2178 * 1024;
        build_coarse_kernel<<<(unsigned)((n + 255) / 256), 256, 0, stream>>>(ids, wte, FX, dummy, X);
    }
    for (int l = 0; l < 2; ++l) run_layer(cl[l], 2178, 1089, true);
    rmsnorm_kernel<<<2178, 256, 0, stream>>>(X, cnorm, XN);
    {
        long n = (long)2048 * 1024;
        gather_final_kernel<<<(unsigned)((n + 255) / 256), 256, 0, stream>>>(XN, XG);
    }
    // logits = XG @ out  (2048 x 32000)
    gemm(XG, 1024, outw, 32000, logits, 32000, nullptr, 2048, 32000, 1024, stream);
}
